// PFA_Mapper_87926570484356
// MI455X (gfx1250) — matrix operand order repaired and checked by reading, not compiled, not run on hardware
//
#include <hip/hip_runtime.h>

// Shapes from the reference
#define CC 10      // channels
#define PP 60000   // pillars
#define KK 32      // points per pillar
#define FF 64      // PFN output features
#define NTILE 3750 // PP / 16  (exact)
#define LDW 33     // padded LDS row stride (odd -> conflict free)

typedef __attribute__((ext_vector_type(16))) _Float16 v16h;
typedef __attribute__((ext_vector_type(8)))  float    v8f;

__device__ __forceinline__ float sigmoidf_(float x) {
    return 1.0f / (1.0f + __expf(-x));
}

// ---------------------------------------------------------------------------
// B-tile pack (16x16x32 f16 WMMA): B is K(32 rows, zero-padded past nrows) x
// 16 cols starting at nbase, from row-major W[nrows][ld].
// Layout: lanes 0-15 hold rows {0..7, 16..23}, lanes 16-31 hold rows {8..15, 24..31};
// VGPR j packs two consecutive rows of that set; lane&15 selects the column.
// ---------------------------------------------------------------------------
__device__ __forceinline__ v16h packB32(const float* __restrict__ W, int ld,
                                        int nrows, int ncols, int nbase, int lane) {
    int n  = lane & 15;
    int kh = lane >> 4;
    v16h b;
#pragma unroll
    for (int j = 0; j < 8; ++j) {
        int r0 = ((j < 4) ? 0 : 16) + kh * 8 + (j & 3) * 2;
        int r1 = r0 + 1;
        int col = nbase + n;
        float f0 = (r0 < nrows && col < ncols) ? W[r0 * ld + col] : 0.0f;
        float f1 = (r1 < nrows && col < ncols) ? W[r1 * ld + col] : 0.0f;
        b[2 * j]     = (_Float16)f0;
        b[2 * j + 1] = (_Float16)f1;
    }
    return b;
}

// fc1 B-tile: B[k=i][n=o] = fc1_w[o*20 + i], o<10, i<20, zero-padded to 32x16
__device__ __forceinline__ v16h packB_fc1(const float* __restrict__ Wfc, int lane) {
    int n  = lane & 15;   // output channel o
    int kh = lane >> 4;
    v16h b;
#pragma unroll
    for (int j = 0; j < 8; ++j) {
        int r0 = ((j < 4) ? 0 : 16) + kh * 8 + (j & 3) * 2;
        int r1 = r0 + 1;
        float f0 = (r0 < 2 * CC && n < CC) ? Wfc[n * 2 * CC + r0] : 0.0f;
        float f1 = (r1 < 2 * CC && n < CC) ? Wfc[n * 2 * CC + r1] : 0.0f;
        b[2 * j]     = (_Float16)f0;
        b[2 * j + 1] = (_Float16)f1;
    }
    return b;
}

// A-tile pack from an LDS [16][LDW] f32 matrix (16 rows x 32 K).
// Documented 16-bit A layout: lanes 0-15 and 16-31 both carry M=0..15;
// lane group 0 covers K {0..7, 16..23}, group 1 covers K {8..15, 24..31}.
__device__ __forceinline__ v16h packA_rows(const float* Y, int lane) {
    int m  = lane & 15;
    int kg = lane >> 4;
    const float* row = Y + m * LDW;
    v16h a;
#pragma unroll
    for (int j = 0; j < 8; ++j) {
        int base = ((j < 4) ? 0 : 16) + kg * 8 + (j & 3) * 2;
        a[2 * j]     = (_Float16)row[base];
        a[2 * j + 1] = (_Float16)row[base + 1];
    }
    return a;
}

// Point-attention layer: YT[16][32] -> sigmoid(relu(YT@W1+b1)@W2+b2) -> PAW
__device__ __forceinline__ void pa_layer(float* YT, float* PAW,
                                         const v16h* bw1, const v16h* bw2,
                                         const float* __restrict__ b1,
                                         const float* __restrict__ b2, int lane) {
    int n = lane & 15, mh = lane >> 4;

    v16h a = packA_rows(YT, lane);
    float bb0 = b1[n], bb1 = b1[16 + n];
    v8f d0 = {}, d1 = {};
    d0 = __builtin_amdgcn_wmma_f32_16x16x32_f16(false, a, false, bw1[0], (short)0, d0, false, false);
    d1 = __builtin_amdgcn_wmma_f32_16x16x32_f16(false, a, false, bw1[1], (short)0, d1, false, false);
    __builtin_amdgcn_wave_barrier();
#pragma unroll
    for (int r = 0; r < 8; ++r) {
        PAW[(mh * 8 + r) * LDW + n]      = fmaxf(d0[r] + bb0, 0.0f);
        PAW[(mh * 8 + r) * LDW + 16 + n] = fmaxf(d1[r] + bb1, 0.0f);
    }
    __builtin_amdgcn_wave_barrier();

    a = packA_rows(PAW, lane);     // reads precede the writes below (in-order LDS)
    bb0 = b2[n]; bb1 = b2[16 + n];
    v8f e0 = {}, e1 = {};
    e0 = __builtin_amdgcn_wmma_f32_16x16x32_f16(false, a, false, bw2[0], (short)0, e0, false, false);
    e1 = __builtin_amdgcn_wmma_f32_16x16x32_f16(false, a, false, bw2[1], (short)0, e1, false, false);
    __builtin_amdgcn_wave_barrier();
#pragma unroll
    for (int r = 0; r < 8; ++r) {
        PAW[(mh * 8 + r) * LDW + n]      = sigmoidf_(e0[r] + bb0);
        PAW[(mh * 8 + r) * LDW + 16 + n] = sigmoidf_(e1[r] + bb1);
    }
    __builtin_amdgcn_wave_barrier();
}

// Channel-attention layer (10x10, VALU): CAW holds Yc[m][c] in, caw[m][c] out.
__device__ __forceinline__ void ca_layer(float* CAW,
                                         const float* __restrict__ w1,
                                         const float* __restrict__ b1,
                                         const float* __restrict__ w2,
                                         const float* __restrict__ b2, int lane) {
    int m = lane & 15;  // lanes 16-31 duplicate lanes 0-15 (harmless)
    float y[CC], h[CC], o[CC];
#pragma unroll
    for (int c = 0; c < CC; ++c) y[c] = CAW[m * 16 + c];
#pragma unroll
    for (int c = 0; c < CC; ++c) {
        float s = b1[c];                 // uniform -> scalar loads
#pragma unroll
        for (int i = 0; i < CC; ++i) s += y[i] * w1[i * CC + c];
        h[c] = fmaxf(s, 0.0f);
    }
#pragma unroll
    for (int c = 0; c < CC; ++c) {
        float s = b2[c];
#pragma unroll
        for (int i = 0; i < CC; ++i) s += h[i] * w2[i * CC + c];
        o[c] = sigmoidf_(s);
    }
    __builtin_amdgcn_wave_barrier();
    if (lane < 16) {
#pragma unroll
        for (int c = 0; c < CC; ++c) CAW[m * 16 + c] = o[c];
    }
    __builtin_amdgcn_wave_barrier();
}

__global__ __launch_bounds__(32) void paca_pfn_kernel(
    const float* __restrict__ x,
    const float* __restrict__ pa1w1, const float* __restrict__ pa1b1,
    const float* __restrict__ pa1w2, const float* __restrict__ pa1b2,
    const float* __restrict__ ca1w1, const float* __restrict__ ca1b1,
    const float* __restrict__ ca1w2, const float* __restrict__ ca1b2,
    const float* __restrict__ fc1w,
    const float* __restrict__ pa2w1, const float* __restrict__ pa2b1,
    const float* __restrict__ pa2w2, const float* __restrict__ pa2b2,
    const float* __restrict__ ca2w1, const float* __restrict__ ca2b1,
    const float* __restrict__ ca2w2, const float* __restrict__ ca2b2,
    const float* __restrict__ pfnw,
    const float* __restrict__ bng, const float* __restrict__ bnb,
    const float* __restrict__ bnm, const float* __restrict__ bnv,
    float* __restrict__ out)
{
    __shared__ float HS[CC * 16 * LDW];  // staged x, then h (in-place), then gated h
    __shared__ float YT[16 * LDW];       // Y / Y2 (max over channels)
    __shared__ float PAW[16 * LDW];      // pa intermediate / pa gate
    __shared__ float CAW[16 * 16];       // Yc / ca gate

    const int lane = threadIdx.x;        // 32-thread block = one wave
    const int p0   = blockIdx.x * 16;    // 16 pillars per tile, NTILE tiles exact

    // ---- pass A: async DMA x tile into LDS (ASYNCcnt path, no VGPR staging) ----
    // GVS mode: mem = SADDR(x) + VADDR(i32 byte offset); per-lane LDS dest is
    // the low 32 bits of the flat __shared__ address (== raw LDS offset).
    {
        unsigned goff0 = ((unsigned)p0 * KK + (unsigned)lane) * 4u;
        for (int c = 0; c < CC; ++c) {
            unsigned gc = goff0 + (unsigned)c * (PP * KK * 4u);
#pragma unroll
            for (int m = 0; m < 16; ++m) {
                unsigned goff = gc + (unsigned)m * (KK * 4u);
                unsigned lds  = (unsigned)(uintptr_t)&HS[(c * 16 + m) * LDW + lane];
                asm volatile("global_load_async_to_lds_b32 %0, %1, %2"
                             :: "v"(lds), "v"(goff), "s"(x)
                             : "memory");
            }
        }
    }

    // ---- overlap with the DMA: pack all weight B-tiles into VGPRs ----
    v16h Bpa1w1[2], Bpa1w2[2], Bpa2w1[2], Bpa2w2[2], Bpfn[4], Bfc;
#pragma unroll
    for (int nt = 0; nt < 2; ++nt) {
        Bpa1w1[nt] = packB32(pa1w1, KK, KK, KK, nt * 16, lane);
        Bpa1w2[nt] = packB32(pa1w2, KK, KK, KK, nt * 16, lane);
        Bpa2w1[nt] = packB32(pa2w1, KK, KK, KK, nt * 16, lane);
        Bpa2w2[nt] = packB32(pa2w2, KK, KK, KK, nt * 16, lane);
    }
#pragma unroll
    for (int nt = 0; nt < 4; ++nt)
        Bpfn[nt] = packB32(pfnw, FF, CC, FF, nt * 16, lane);
    Bfc = packB_fc1(fc1w, lane);

    // ---- per-lane BN scale/bias (f = nt*16 + (lane&15)) ----
    float bnscale[4], bnbias[4];
    {
        int n = lane & 15;
#pragma unroll
        for (int nt = 0; nt < 4; ++nt) {
            int f = nt * 16 + n;
            float sc = bng[f] * rsqrtf(bnv[f] + 1e-3f);
            bnscale[nt] = sc;
            bnbias[nt]  = bnb[f] - bnm[f] * sc;
        }
    }

    // ---- DMA done? then compute Y = max over channels from LDS ----
    asm volatile("s_wait_asynccnt 0" ::: "memory");
    __builtin_amdgcn_wave_barrier();

    for (int m = 0; m < 16; ++m) {
        float ym = -3.4e38f;
#pragma unroll
        for (int c = 0; c < CC; ++c)
            ym = fmaxf(ym, HS[(c * 16 + m) * LDW + lane]);
        YT[m * LDW + lane] = ym;
    }
    __builtin_amdgcn_wave_barrier();

    // ---- Yc = max over K : 160 (c,m) rows, 5 per lane ----
#pragma unroll
    for (int q = 0; q < 5; ++q) {
        int pair = q * 32 + lane;            // pair == c*16 + m
        const float* row = &HS[pair * LDW];
        float v = row[0];
#pragma unroll
        for (int k = 1; k < KK; ++k) v = fmaxf(v, row[k]);
        CAW[(pair & 15) * 16 + (pair >> 4)] = v;   // CAW[m][c]
    }
    __builtin_amdgcn_wave_barrier();

    // ---- PACA layer 1 attention weights ----
    ca_layer(CAW, ca1w1, ca1b1, ca1w2, ca1b2, lane);
    pa_layer(YT, PAW, Bpa1w1, Bpa1w2, pa1b1, pa1b2, lane);

    // ---- fc1: 1x1 conv over concat([x, x*paw*caw]) via WMMA, in-place in HS ----
    {
        int rowi = lane & 15, grp = lane >> 4;
        int n = lane & 15, mh = lane >> 4;
        for (int t = 0; t < 32; ++t) {
            int m  = t >> 1;
            int kb = (t & 1) * 16;
            int k  = kb + rowi;
            float pw = PAW[m * LDW + k];
            v16h a;
#pragma unroll
            for (int j = 0; j < 8; ++j) {
                int i0 = ((j < 4) ? 0 : 16) + grp * 8 + (j & 3) * 2;
#pragma unroll
                for (int half = 0; half < 2; ++half) {
                    int i = i0 + half;
                    float f;
                    if (i < CC) {
                        f = HS[(i * 16 + m) * LDW + k];
                    } else if (i < 2 * CC) {
                        int c = i - CC;
                        f = HS[(c * 16 + m) * LDW + k] * pw * CAW[m * 16 + c];
                    } else {
                        f = 0.0f;
                    }
                    a[2 * j + half] = (_Float16)f;
                }
            }
            v8f d = {};
            d = __builtin_amdgcn_wmma_f32_16x16x32_f16(false, a, false, Bfc, (short)0, d, false, false);
            __builtin_amdgcn_wave_barrier();
            if (n < CC) {
#pragma unroll
                for (int r = 0; r < 8; ++r)
                    HS[(n * 16 + m) * LDW + kb + mh * 8 + r] = d[r];
            }
            __builtin_amdgcn_wave_barrier();
        }
    }

    // ---- Y2 = max over channels of h ----
    for (int m = 0; m < 16; ++m) {
        float ym = -3.4e38f;
#pragma unroll
        for (int c = 0; c < CC; ++c)
            ym = fmaxf(ym, HS[(c * 16 + m) * LDW + lane]);
        YT[m * LDW + lane] = ym;
    }
    __builtin_amdgcn_wave_barrier();

    // ---- Yc2 ----
#pragma unroll
    for (int q = 0; q < 5; ++q) {
        int pair = q * 32 + lane;
        const float* row = &HS[pair * LDW];
        float v = row[0];
#pragma unroll
        for (int k = 1; k < KK; ++k) v = fmaxf(v, row[k]);
        CAW[(pair & 15) * 16 + (pair >> 4)] = v;
    }
    __builtin_amdgcn_wave_barrier();

    // ---- PACA layer 2 attention weights ----
    ca_layer(CAW, ca2w1, ca2b1, ca2w2, ca2b2, lane);
    pa_layer(YT, PAW, Bpa2w1, Bpa2w2, pa2b1, pa2b2, lane);

    // ---- gate h in place: h *= paw2 * caw2 ----
    for (int m = 0; m < 16; ++m) {
        float pw = PAW[m * LDW + lane];
#pragma unroll
        for (int c = 0; c < CC; ++c) {
            int idx = (c * 16 + m) * LDW + lane;
            HS[idx] = HS[idx] * pw * CAW[m * 16 + c];
        }
    }
    __builtin_amdgcn_wave_barrier();

    // ---- PFN: (per pillar) 32x10 @ 10x64 via WMMA, BN + ReLU + max over points ----
    {
        int rowi = lane & 15, grp = lane >> 4;
        for (int m = 0; m < 16; ++m) {
            float runmax[4] = {0.0f, 0.0f, 0.0f, 0.0f};  // post-ReLU values are >= 0
#pragma unroll
            for (int kt = 0; kt < 2; ++kt) {
                int k = kt * 16 + rowi;
                v16h a;
#pragma unroll
                for (int j = 0; j < 8; ++j) {
                    int i0 = ((j < 4) ? 0 : 16) + grp * 8 + (j & 3) * 2;
                    float f0 = (i0     < CC) ? HS[(i0 * 16 + m) * LDW + k]       : 0.0f;
                    float f1 = (i0 + 1 < CC) ? HS[((i0 + 1) * 16 + m) * LDW + k] : 0.0f;
                    a[2 * j]     = (_Float16)f0;
                    a[2 * j + 1] = (_Float16)f1;
                }
#pragma unroll
                for (int nt = 0; nt < 4; ++nt) {
                    v8f d = {};
                    d = __builtin_amdgcn_wmma_f32_16x16x32_f16(false, a, false, Bpfn[nt],
                                                               (short)0, d, false, false);
                    float mx = 0.0f;
#pragma unroll
                    for (int r = 0; r < 8; ++r) {
                        float z = fmaxf(d[r] * bnscale[nt] + bnbias[nt], 0.0f);
                        mx = fmaxf(mx, z);
                    }
                    mx = fmaxf(mx, __shfl_xor(mx, 16, 32));  // combine row halves
                    runmax[nt] = fmaxf(runmax[nt], mx);
                }
            }
            if (lane < 16) {
#pragma unroll
                for (int nt = 0; nt < 4; ++nt)
                    out[(p0 + m) * FF + nt * 16 + lane] = runmax[nt];
            }
        }
    }
}

extern "C" void kernel_launch(void* const* d_in, const int* in_sizes, int n_in,
                              void* d_out, int out_size, void* d_ws, size_t ws_size,
                              hipStream_t stream) {
    (void)in_sizes; (void)n_in; (void)d_ws; (void)ws_size; (void)out_size;
    const float* x     = (const float*)d_in[0];
    const float* pa1w1 = (const float*)d_in[1];
    const float* pa1b1 = (const float*)d_in[2];
    const float* pa1w2 = (const float*)d_in[3];
    const float* pa1b2 = (const float*)d_in[4];
    const float* ca1w1 = (const float*)d_in[5];
    const float* ca1b1 = (const float*)d_in[6];
    const float* ca1w2 = (const float*)d_in[7];
    const float* ca1b2 = (const float*)d_in[8];
    const float* fc1w  = (const float*)d_in[9];
    const float* pa2w1 = (const float*)d_in[10];
    const float* pa2b1 = (const float*)d_in[11];
    const float* pa2w2 = (const float*)d_in[12];
    const float* pa2b2 = (const float*)d_in[13];
    const float* ca2w1 = (const float*)d_in[14];
    const float* ca2b1 = (const float*)d_in[15];
    const float* ca2w2 = (const float*)d_in[16];
    const float* ca2b2 = (const float*)d_in[17];
    const float* pfnw  = (const float*)d_in[18];
    const float* bng   = (const float*)d_in[19];
    const float* bnb   = (const float*)d_in[20];
    const float* bnm   = (const float*)d_in[21];
    const float* bnv   = (const float*)d_in[22];
    float* outp = (float*)d_out;

    paca_pfn_kernel<<<dim3(NTILE), dim3(32), 0, stream>>>(
        x, pa1w1, pa1b1, pa1w2, pa1b2, ca1w1, ca1b1, ca1w2, ca1b2, fc1w,
        pa2w1, pa2b1, pa2w2, pa2b2, ca2w1, ca2b1, ca2w2, ca2b2,
        pfnw, bng, bnb, bnm, bnv, outp);
}
